// Attention_19868518711477
// MI455X (gfx1250) — compile-verified, matrix operand order checked by reading, not run on hardware
//
#include <hip/hip_runtime.h>
#include <hip/hip_bf16.h>

typedef __attribute__((ext_vector_type(16))) _Float16 v16h;
typedef __attribute__((ext_vector_type(8)))  _Float16 v8h;
typedef __attribute__((ext_vector_type(8)))  float    v8f;

#define BATCH 2
#define SEQ   2048
#define DIM   1024
#define HEADS 16
#define DH    64
#define INNER 1024
#define E3    3072
#define MROWS (BATCH*SEQ)   /* 4096 */
#define QK_SCALE 0.125f     /* 64^-0.5 */

// ---------------------------------------------------------------------------
// WMMA helper: D = A(16x32 f16) * B(32x16 f16) + C(16x16 f32)
// ---------------------------------------------------------------------------
__device__ __forceinline__ v8f wmma_f16(v16h a, v16h b, v8f c) {
#if defined(__HIP_DEVICE_COMPILE__)
  return __builtin_amdgcn_wmma_f32_16x16x32_f16(false, a, false, b,
                                                (short)0, c, false, false);
#else
  return c;
#endif
}

// Load a 16x32 f16 fragment whose *contraction* dim is contiguous in memory.
//   lanes 0-15:  row = lane,    K = {0..7, 16..23}
//   lanes 16-31: row = lane-16, K = {8..15, 24..31}
__device__ __forceinline__ v16h load_frag16(const _Float16* base, int ld) {
  int lane = threadIdx.x & 31;
  int row  = lane & 15;
  int koff = (lane >> 4) << 3;          // 0 or 8
  const _Float16* p = base + (size_t)row * ld + koff;
  v8h lo = *(const v8h*)(p);
  v8h hi = *(const v8h*)(p + 16);
  v16h r;
#pragma unroll
  for (int i = 0; i < 8; ++i) { r[i] = lo[i]; r[8 + i] = hi[i]; }
  return r;
}

// Load one k-step worth of fragments: 2 A-frags + 4 B-frags.
__device__ __forceinline__ void load_group(const _Float16* A0, const _Float16* A1,
                                           const _Float16* Wt, int kq, int ld,
                                           v16h& a0, v16h& a1, v16h (&b)[4]) {
  a0 = load_frag16(A0 + kq, ld);
  a1 = load_frag16(A1 + kq, ld);
#pragma unroll
  for (int t = 0; t < 4; ++t)
    b[t] = load_frag16(Wt + (size_t)(t * 16) * ld + kq, ld);
}

// ---------------------------------------------------------------------------
// Kernel 1: fp32 -> fp16 conversion (grid-stride)
// ---------------------------------------------------------------------------
__global__ void cvt_f32_f16(const float* __restrict__ src,
                            _Float16* __restrict__ dst, int n) {
  int i = blockIdx.x * blockDim.x + threadIdx.x;
  int stride = gridDim.x * blockDim.x;
  for (; i < n; i += stride) dst[i] = (_Float16)src[i];
}

// ---------------------------------------------------------------------------
// Kernel 2: QKV projection.  qkv[m,e] = sum_k xh[m,k] * wqkv[e,k]
// Wave tile: 32(M) x 64(E).  2x-unrolled k-loop with structural ping-pong
// fragment sets (no end-of-iteration register rotation -> no v_mov walls):
//   load B(k+32) ; WMMA A(k) ; load A(k+64) ; WMMA B(k+32)
// Epilogue routes e into q (scaled), k, or v; V stored TRANSPOSED
// as vt[bh][dh][n] so attention P*V B-frags are contiguous.
// ---------------------------------------------------------------------------
__global__ __launch_bounds__(256) void qkv_gemm(const _Float16* __restrict__ xh,
                                                const _Float16* __restrict__ wh,
                                                _Float16* __restrict__ q,
                                                _Float16* __restrict__ k,
                                                _Float16* __restrict__ vt) {
  int lane = threadIdx.x;
  int wave = threadIdx.y;
  int m0 = blockIdx.y * 256 + wave * 32;   // 32 rows per wave
  int e0 = blockIdx.x * 64;

  const _Float16* Arow0 = xh + (size_t)m0 * DIM;
  const _Float16* Arow1 = xh + (size_t)(m0 + 16) * DIM;
  const _Float16* Wt    = wh + (size_t)e0 * DIM;

  v8f acc[2][4] = {};

  v16h a0A, a1A, bA[4];     // ping
  v16h a0B, a1B, bB[4];     // pong
  load_group(Arow0, Arow1, Wt, 0, DIM, a0A, a1A, bA);

  for (int k0 = 0; k0 < DIM; k0 += 64) {
    load_group(Arow0, Arow1, Wt, k0 + 32, DIM, a0B, a1B, bB);
#pragma unroll
    for (int t = 0; t < 4; ++t) {
      acc[0][t] = wmma_f16(a0A, bA[t], acc[0][t]);
      acc[1][t] = wmma_f16(a1A, bA[t], acc[1][t]);
    }
    int kn = (k0 + 64 < DIM) ? (k0 + 64) : 0;   // tail: harmless reload of k=0
    load_group(Arow0, Arow1, Wt, kn, DIM, a0A, a1A, bA);
#pragma unroll
    for (int t = 0; t < 4; ++t) {
      acc[0][t] = wmma_f16(a0B, bB[t], acc[0][t]);
      acc[1][t] = wmma_f16(a1B, bB[t], acc[1][t]);
    }
  }

  // C layout: VGPR r -> row (r + 8*(lane>=16)), col = lane&15
  int col   = lane & 15;
  int rbase = (lane >> 4) << 3;
#pragma unroll
  for (int rf = 0; rf < 2; ++rf) {
#pragma unroll
    for (int t = 0; t < 4; ++t) {
      int e = e0 + t * 16 + col;
      int which = e >> 10;                 // 0=q, 1=k, 2=v
      int r = e & 1023;
      int h = r >> 6;
      int d = r & 63;
#pragma unroll
      for (int rr = 0; rr < 8; ++rr) {
        int m  = m0 + rf * 16 + rbase + rr;
        int b_ = m >> 11;                  // / 2048
        int n  = m & 2047;
        int bh = b_ * HEADS + h;
        float val = acc[rf][t][rr];
        if (which == 0)
          q[((size_t)bh * SEQ + n) * DH + d] = (_Float16)(val * QK_SCALE);
        else if (which == 1)
          k[((size_t)bh * SEQ + n) * DH + d] = (_Float16)val;
        else
          vt[((size_t)bh * DH + d) * SEQ + n] = (_Float16)val;   // transposed
      }
    }
  }
}

// ---------------------------------------------------------------------------
// Kernel 3: flash attention.  One wave owns a 16-row query tile of one head.
// Per 32-key chunk: K and V fragment loads are BOTH issued up front (loads
// retire in order, so S-WMMAs only wait for the K portion while V loads hide
// under the softmax VALU stretch), then online softmax, P via per-wave LDS
// round-trip (C-layout -> A-layout), then O += P*V.
// ---------------------------------------------------------------------------
__global__ __launch_bounds__(256) void flash_attn(const _Float16* __restrict__ q,
                                                  const _Float16* __restrict__ kk,
                                                  const _Float16* __restrict__ vt,
                                                  _Float16* __restrict__ oh) {
  __shared__ __align__(16) _Float16 pst[8][16][32];   // per-wave P staging
  int lane = threadIdx.x;
  int wave = threadIdx.y;
  int bh = blockIdx.x;                       // 0..31 (b*16+h)
  int mt = blockIdx.y * 8 + wave;            // 0..127 query tile
  int b_ = bh >> 4, h = bh & 15;

  const _Float16* Q = q  + ((size_t)bh * SEQ + (size_t)mt * 16) * DH;
  const _Float16* K = kk + (size_t)bh * SEQ * DH;
  const _Float16* V = vt + (size_t)bh * DH * SEQ;    // [64][2048]

  v16h qa0 = load_frag16(Q, DH);             // dh 0..31
  v16h qa1 = load_frag16(Q + 32, DH);        // dh 32..63

  v8f o[4] = {};
  float mrow[8], lrow[8];
#pragma unroll
  for (int r = 0; r < 8; ++r) { mrow[r] = -3.0e38f; lrow[r] = 0.f; }

  int col   = lane & 15;
  int rbase = (lane >> 4) << 3;
  int koff  = (lane >> 4) << 3;

  for (int j0 = 0; j0 < SEQ; j0 += 32) {
    __builtin_prefetch(K + (size_t)(j0 + 32) * DH, 0, 0);
    __builtin_prefetch(V + j0 + 32, 0, 0);

    // ---- issue K frags, then V frags (V stays outstanding through softmax)
    v16h kb00 = load_frag16(K + (size_t)j0 * DH,             DH);
    v16h kb01 = load_frag16(K + (size_t)j0 * DH + 32,        DH);
    v16h kb10 = load_frag16(K + (size_t)(j0 + 16) * DH,      DH);
    v16h kb11 = load_frag16(K + (size_t)(j0 + 16) * DH + 32, DH);
    v16h vb[4];
#pragma unroll
    for (int t = 0; t < 4; ++t)
      vb[t] = load_frag16(V + (size_t)(t * 16) * SEQ + j0, SEQ);

    // ---- S = Q * K^T
    v8f s0 = {}, s1 = {};
    s0 = wmma_f16(qa0, kb00, s0);
    s0 = wmma_f16(qa1, kb01, s0);
    s1 = wmma_f16(qa0, kb10, s1);
    s1 = wmma_f16(qa1, kb11, s1);

    // ---- online softmax (rows per-VGPR, cols across a 16-lane half)
#pragma unroll
    for (int r = 0; r < 8; ++r) {
      float cm = fmaxf(s0[r], s1[r]);
#pragma unroll
      for (int off = 1; off <= 8; off <<= 1)
        cm = fmaxf(cm, __shfl_xor(cm, off, 32));
      float mn = fmaxf(mrow[r], cm);
      float alpha = __expf(mrow[r] - mn);
      mrow[r] = mn;
      float p0 = __expf(s0[r] - mn);
      float p1 = __expf(s1[r] - mn);
      s0[r] = p0; s1[r] = p1;
      float rs = p0 + p1;
#pragma unroll
      for (int off = 1; off <= 8; off <<= 1)
        rs += __shfl_xor(rs, off, 32);
      lrow[r] = lrow[r] * alpha + rs;
      o[0][r] *= alpha; o[1][r] *= alpha;
      o[2][r] *= alpha; o[3][r] *= alpha;
    }

    // ---- P: C-layout -> LDS row-major [16][32] -> A-layout fragment
#pragma unroll
    for (int r = 0; r < 8; ++r) {
      pst[wave][rbase + r][col]      = (_Float16)s0[r];
      pst[wave][rbase + r][col + 16] = (_Float16)s1[r];
    }
    // same-wave LDS ops are in-order (ISA ch.11): no barrier needed
    v8h plo = *(const v8h*)&pst[wave][col][koff];
    v8h phi = *(const v8h*)&pst[wave][col][koff + 16];
    v16h pa;
#pragma unroll
    for (int i = 0; i < 8; ++i) { pa[i] = plo[i]; pa[8 + i] = phi[i]; }

    // ---- O += P * V  (V frags already resident)
#pragma unroll
    for (int t = 0; t < 4; ++t)
      o[t] = wmma_f16(pa, vb[t], o[t]);
  }

  // ---- epilogue: normalize by row sum, store f16 at [b, n, h*64+d]
#pragma unroll
  for (int r = 0; r < 8; ++r) {
    float inv = 1.0f / lrow[r];
    int n = mt * 16 + rbase + r;
    size_t rowbase = ((size_t)b_ * SEQ + n) * INNER + (size_t)h * DH;
#pragma unroll
    for (int t = 0; t < 4; ++t)
      oh[rowbase + t * 16 + col] = (_Float16)(o[t][r] * inv);
  }
}

// ---------------------------------------------------------------------------
// Kernel 4: output projection.  out[m,d] = sum_e oh[m,e] * wout[d,e]  (fp32)
// Same 32x64 ping-pong pipelined wave tile as qkv_gemm.
// ---------------------------------------------------------------------------
__global__ __launch_bounds__(256) void out_gemm(const _Float16* __restrict__ oh,
                                                const _Float16* __restrict__ wh,
                                                float* __restrict__ out) {
  int lane = threadIdx.x;
  int wave = threadIdx.y;
  int m0 = blockIdx.y * 256 + wave * 32;
  int d0 = blockIdx.x * 64;

  const _Float16* Arow0 = oh + (size_t)m0 * INNER;
  const _Float16* Arow1 = oh + (size_t)(m0 + 16) * INNER;
  const _Float16* Wt    = wh + (size_t)d0 * INNER;

  v8f acc[2][4] = {};

  v16h a0A, a1A, bA[4];
  v16h a0B, a1B, bB[4];
  load_group(Arow0, Arow1, Wt, 0, INNER, a0A, a1A, bA);

  for (int k0 = 0; k0 < INNER; k0 += 64) {
    load_group(Arow0, Arow1, Wt, k0 + 32, INNER, a0B, a1B, bB);
#pragma unroll
    for (int t = 0; t < 4; ++t) {
      acc[0][t] = wmma_f16(a0A, bA[t], acc[0][t]);
      acc[1][t] = wmma_f16(a1A, bA[t], acc[1][t]);
    }
    int kn = (k0 + 64 < INNER) ? (k0 + 64) : 0;
    load_group(Arow0, Arow1, Wt, kn, INNER, a0A, a1A, bA);
#pragma unroll
    for (int t = 0; t < 4; ++t) {
      acc[0][t] = wmma_f16(a0B, bB[t], acc[0][t]);
      acc[1][t] = wmma_f16(a1B, bB[t], acc[1][t]);
    }
  }

  int col   = lane & 15;
  int rbase = (lane >> 4) << 3;
#pragma unroll
  for (int rf = 0; rf < 2; ++rf)
#pragma unroll
    for (int t = 0; t < 4; ++t)
#pragma unroll
      for (int rr = 0; rr < 8; ++rr) {
        int m = m0 + rf * 16 + rbase + rr;
        out[(size_t)m * DIM + d0 + t * 16 + col] = acc[rf][t][rr];
      }
}

// ---------------------------------------------------------------------------
// Launch: convert -> qkv gemm -> flash attention -> output gemm
// Workspace layout (f16 elements), total 48 MB:
//   xh[4096*1024] wqkv[3072*1024] wout[1024*1024]
//   q[32*2048*64] k[32*2048*64] vt[32*64*2048] oh[4096*1024]
// ---------------------------------------------------------------------------
extern "C" void kernel_launch(void* const* d_in, const int* in_sizes, int n_in,
                              void* d_out, int out_size, void* d_ws, size_t ws_size,
                              hipStream_t stream) {
  const float* x     = (const float*)d_in[0];
  const float* w_qkv = (const float*)d_in[1];
  const float* w_out = (const float*)d_in[2];
  float* out = (float*)d_out;

  _Float16* ws    = (_Float16*)d_ws;
  _Float16* xh    = ws;
  _Float16* wqkvh = xh    + (size_t)MROWS * DIM;      // +4M
  _Float16* wouth = wqkvh + (size_t)E3 * DIM;         // +3M
  _Float16* qb    = wouth + (size_t)DIM * INNER;      // +1M
  _Float16* kb    = qb    + (size_t)MROWS * INNER;    // +4M
  _Float16* vtb   = kb    + (size_t)MROWS * INNER;    // +4M
  _Float16* ohb   = vtb   + (size_t)MROWS * INNER;    // +4M

  cvt_f32_f16<<<1024, 256, 0, stream>>>(x,     xh,    MROWS * DIM);
  cvt_f32_f16<<<1024, 256, 0, stream>>>(w_qkv, wqkvh, E3 * DIM);
  cvt_f32_f16<<<1024, 256, 0, stream>>>(w_out, wouth, DIM * INNER);

  qkv_gemm<<<dim3(E3 / 64, MROWS / 256), dim3(32, 8), 0, stream>>>(
      xh, wqkvh, qb, kb, vtb);

  flash_attn<<<dim3(BATCH * HEADS, SEQ / (16 * 8)), dim3(32, 8), 0, stream>>>(
      qb, kb, vtb, ohb);

  out_gemm<<<dim3(DIM / 64, MROWS / 256), dim3(32, 8), 0, stream>>>(
      ohb, wouth, out);
}
